// LlmcFp8Linear_56375740727544
// MI455X (gfx1250) — 8-bit matrix operand order repaired and checked by reading, compiled, not run on hardware
//
#include <hip/hip_runtime.h>
#include <hip/hip_bf16.h>
#include <stdint.h>

typedef __attribute__((ext_vector_type(16))) int   v16i;
typedef __attribute__((ext_vector_type(8)))  float v8f;

#define M_DIM 8192
#define K_DIM 4096
#define N_DIM 12288
#define BS    128
#define KB    (K_DIM / BS)

__device__ __forceinline__ unsigned int f32_to_e4m3_manual(float x) {
    unsigned int s = (__float_as_uint(x) >> 31) << 7;
    float ax = fabsf(x);
    if (!(ax == ax)) return s | 0x7F;
    ax = fminf(ax, 448.0f);
    if (ax < 0.0009765625f) return s;
    int e; float f = frexpf(ax, &e);
    int E = e + 6;
    unsigned int code;
    if (E <= 0) {
        code = (unsigned int)rintf(ax * 512.0f);
    } else {
        int m = (int)rintf((2.0f * f - 1.0f) * 8.0f);
        if (m == 8) { m = 0; E += 1; }
        if (E > 15) { E = 15; m = 6; }
        code = ((unsigned int)E << 3) | (unsigned int)m;
    }
    return s | code;
}

__device__ __forceinline__ unsigned int pack4_fp8(float a, float b, float c, float d) {
#if __has_builtin(__builtin_amdgcn_cvt_pk_fp8_f32)
    unsigned int lo = (unsigned int)__builtin_amdgcn_cvt_pk_fp8_f32(a, b, 0, false);
    return (unsigned int)__builtin_amdgcn_cvt_pk_fp8_f32(c, d, (int)lo, true);
#else
    return f32_to_e4m3_manual(a) | (f32_to_e4m3_manual(b) << 8) |
           (f32_to_e4m3_manual(c) << 16) | (f32_to_e4m3_manual(d) << 24);
#endif
}

__global__ __launch_bounds__(256) void quant_x_kernel(
    const float* __restrict__ x, unsigned char* __restrict__ xq,
    float* __restrict__ sxT) {
    const int lane = threadIdx.x & 31;
    const int wave = threadIdx.x >> 5;
    const int task = blockIdx.x * 8 + wave;
    const int m  = task / KB;
    const int kb = task - m * KB;
    const size_t base = (size_t)m * K_DIM + (size_t)kb * BS;

    float4 v = *reinterpret_cast<const float4*>(x + base + lane * 4);
    float amax = fmaxf(fmaxf(fabsf(v.x), fabsf(v.y)), fmaxf(fabsf(v.z), fabsf(v.w)));
#pragma unroll
    for (int off = 16; off > 0; off >>= 1)
        amax = fmaxf(amax, __shfl_xor(amax, off, 32));

    float s   = fmaxf(amax * (1.0f / 448.0f), 1e-12f);
    float inv = 1.0f / s;
    unsigned int packed = pack4_fp8(v.x * inv, v.y * inv, v.z * inv, v.w * inv);

    const int p0 = ((lane >> 2) & 1) * 64 + (lane >> 3) * 16 + (lane & 3) * 4;
    *reinterpret_cast<unsigned int*>(xq + base + p0) = packed;
    if (lane == 0) sxT[(size_t)kb * M_DIM + m] = s;
}

__global__ __launch_bounds__(256) void conv_w_kernel(
    const float* __restrict__ w, unsigned char* __restrict__ wq) {
    const size_t i    = (size_t)blockIdx.x * 256 + threadIdx.x;
    const size_t base = i * 4;
    float4 v = *reinterpret_cast<const float4*>(w + base);
    *reinterpret_cast<unsigned int*>(wq + base) = pack4_fp8(v.x, v.y, v.z, v.w);
}

__global__ __launch_bounds__(256) void gemm_fp8_kernel(
    const unsigned char* __restrict__ xq, const unsigned char* __restrict__ wq,
    const float* __restrict__ sxT, const float* __restrict__ wscale,
    const float* __restrict__ bias, __hip_bfloat16* __restrict__ out) {
    const int lane = threadIdx.x & 31;
    const int wave = threadIdx.x >> 5;
    const int hl   = lane >> 4;
    const int l16  = lane & 15;

    const int n0_wg = blockIdx.x * 128;
    const int m0_wg = blockIdx.y * 128;
    const int wm = wave & 3;
    const int wn = wave >> 2;
    const int m0 = m0_wg + wm * 32;
    const int n0 = n0_wg + wn * 64;
    const int nblk = blockIdx.x;

    v8f acc[2][4];
#pragma unroll
    for (int a = 0; a < 2; ++a)
#pragma unroll
        for (int b = 0; b < 4; ++b)
            acc[a][b] = (v8f){};

    for (int kb = 0; kb < KB; ++kb) {
        const size_t koff = (size_t)kb * BS;
        const float sW = wscale[nblk * KB + kb];

        v16i A[2];
#pragma unroll
        for (int mt = 0; mt < 2; ++mt) {
            const unsigned char* ap =
                xq + (size_t)(m0 + mt * 16 + l16) * K_DIM + koff + hl * 64;
#pragma unroll
            for (int q = 0; q < 4; ++q) {
                uint4 ld = *reinterpret_cast<const uint4*>(ap + 16 * q);
                A[mt][4 * q]     = (int)ld.x;
                A[mt][4 * q + 1] = (int)ld.y;
                A[mt][4 * q + 2] = (int)ld.z;
                A[mt][4 * q + 3] = (int)ld.w;
            }
        }

        v16i B[4];
#pragma unroll
        for (int nt = 0; nt < 4; ++nt) {
            const unsigned char* bp =
                wq + (size_t)(n0 + nt * 16 + l16) * K_DIM + koff + hl * 16;
#pragma unroll
            for (int c = 0; c < 4; ++c) {
                uint4 ld = *reinterpret_cast<const uint4*>(bp + 32 * c);
                B[nt][4 * c]     = (int)ld.x;
                B[nt][4 * c + 1] = (int)ld.y;
                B[nt][4 * c + 2] = (int)ld.z;
                B[nt][4 * c + 3] = (int)ld.w;
            }
        }

        float coeff[2][8];
#pragma unroll
        for (int mt = 0; mt < 2; ++mt) {
            const float* sp = sxT + (size_t)kb * M_DIM + (m0 + mt * 16 + hl * 8);
            float4 c0 = *reinterpret_cast<const float4*>(sp);
            float4 c1 = *reinterpret_cast<const float4*>(sp + 4);
            coeff[mt][0] = c0.x * sW; coeff[mt][1] = c0.y * sW;
            coeff[mt][2] = c0.z * sW; coeff[mt][3] = c0.w * sW;
            coeff[mt][4] = c1.x * sW; coeff[mt][5] = c1.y * sW;
            coeff[mt][6] = c1.z * sW; coeff[mt][7] = c1.w * sW;
        }

#pragma unroll
        for (int mt = 0; mt < 2; ++mt)
#pragma unroll
            for (int nt = 0; nt < 4; ++nt) {
                v8f zero = (v8f){};
                v8f d = __builtin_amdgcn_wmma_f32_16x16x128_fp8_fp8(
                    A[mt], B[nt], (short)0, zero, false, false);
#pragma unroll
                for (int i = 0; i < 8; ++i)
                    acc[mt][nt][i] += d[i] * coeff[mt][i];
            }
    }

#pragma unroll
    for (int nt = 0; nt < 4; ++nt) {
        const int col = n0 + nt * 16 + l16;
        const float bv = bias[col];
#pragma unroll
        for (int mt = 0; mt < 2; ++mt)
#pragma unroll
            for (int i = 0; i < 8; ++i) {
                const int row = m0 + mt * 16 + hl * 8 + i;
                out[(size_t)row * N_DIM + col] =
                    __float2bfloat16(acc[mt][nt][i] + bv);
            }
    }
}

extern "C" void kernel_launch(void* const* d_in, const int* in_sizes, int n_in,
                              void* d_out, int out_size, void* d_ws, size_t ws_size,
                              hipStream_t stream) {
    const float* x      = (const float*)d_in[0];
    const float* w      = (const float*)d_in[1];
    const float* wscale = (const float*)d_in[2];
    const float* bias   = (const float*)d_in[3];

    unsigned char* xq  = (unsigned char*)d_ws;
    unsigned char* wq  = xq + (size_t)M_DIM * K_DIM;
    float*         sxT = (float*)(wq + (size_t)N_DIM * K_DIM);
    __hip_bfloat16* out = (__hip_bfloat16*)d_out;

    quant_x_kernel<<<(M_DIM * KB) / 8, 256, 0, stream>>>(x, xq, sxT);
    conv_w_kernel<<<((size_t)N_DIM * K_DIM) / 1024, 256, 0, stream>>>(w, wq);
    dim3 grid(N_DIM / 128, M_DIM / 128);
    gemm_fp8_kernel<<<grid, 256, 0, stream>>>(xq, wq, sxT, wscale, bias, out);
}
